// GATEncoder_55920474194402
// MI455X (gfx1250) — matrix operand order repaired and checked by reading, not compiled, not run on hardware
//
#include <hip/hip_runtime.h>
#include <hip/hip_bf16.h>

typedef __attribute__((ext_vector_type(16))) _Float16 v16h;
typedef __attribute__((ext_vector_type(8)))  float    v8f;

#define FIN 128   // input feature dim for every layer (128 = H*F_HID)
#define FPH 64    // features per head (F_HID == F_OUT == 64)

// ---------- helpers ----------
__device__ __forceinline__ unsigned fenc(float f) {
    unsigned u = __float_as_uint(f);
    return (u & 0x80000000u) ? ~u : (u | 0x80000000u);
}
__device__ __forceinline__ float fdec(unsigned u) {
    u = (u & 0x80000000u) ? (u & 0x7FFFFFFFu) : ~u;
    return __uint_as_float(u);
}
__device__ __forceinline__ float lrelu(float v) { return v > 0.f ? v : 0.2f * v; }
__device__ __forceinline__ unsigned pack2h(float x, float y) {
    unsigned short h0 = __builtin_bit_cast(unsigned short, (_Float16)x);
    unsigned short h1 = __builtin_bit_cast(unsigned short, (_Float16)y);
    return (unsigned)h0 | ((unsigned)h1 << 16);
}

// ---------- pack W fp32 -> fragment-ordered fp16 ----------
// Wp linear index = ((ct*4 + kt)*32 + lane)*16 + i  (B fragment of v_wmma_f32_16x16x32_f16)
//   k = kt*32 + i + (lane&16 ? 16 : 0), col = ct*16 + (lane&15)
template <int FOUT>
__global__ void k_pack_w(const float* __restrict__ W, _Float16* __restrict__ Wp) {
    int idx = blockIdx.x * blockDim.x + threadIdx.x;
    if (idx >= FOUT * FIN) return;
    int i    = idx & 15;
    int lane = (idx >> 4) & 31;
    int kt   = (idx >> 9) & 3;
    int ct   = idx >> 11;
    int k    = kt * 32 + i + ((lane & 16) ? 16 : 0);
    int col  = ct * 16 + (lane & 15);
    Wp[idx] = (_Float16)W[k * FOUT + col];
}

// ---------- GEMM: H = X[n,128] @ W[128,FOUT] via v_wmma_f32_16x16x32_f16 ----------
// block: 32 * (FOUT/16) threads; one 16-row M tile per block, one 16-col tile per wave.
// A tile staged in LDS pre-converted to f16 in fragment order: [kt][lane][i].
template <int FOUT>
__global__ void k_gemm(const float* __restrict__ X, const _Float16* __restrict__ Wp,
                       float* __restrict__ Hout, int n) {
    __shared__ __align__(16) unsigned sAu[4 * 32 * 8];   // 4KB: 4 kt x 32 lane x 16 half
    const int mtile = blockIdx.x;
    const int tid   = threadIdx.x;

    // stage + convert A fragments (u32 = packed half pair)
    for (int idx = tid; idx < 4 * 32 * 8; idx += blockDim.x) {
        int j    = idx & 7;
        int lane = (idx >> 3) & 31;
        int kt   = idx >> 8;
        int row  = mtile * 16 + (lane & 15);
        // fragment elems i0=2j,i1=2j+1 -> k0 = 2j + (lane&16?16:0) + kt*32
        int k0   = 2 * j + ((lane & 16) ? 16 : 0) + kt * 32;
        float x0 = 0.f, x1 = 0.f;
        if (row < n) {
            const float* rp = X + row * FIN + k0;   // n*FIN fits int32
            x0 = rp[0]; x1 = rp[1];
        }
        sAu[idx] = pack2h(x0, x1);
    }
    __syncthreads();

    const int wave = tid >> 5;
    const int lane = tid & 31;
    const int col  = wave * 16 + (lane & 15);
    const _Float16* sAh = (const _Float16*)sAu;

    v8f c = {};
#pragma unroll
    for (int kt = 0; kt < 4; ++kt) {
        v16h a = *(const v16h*)(sAh + (kt * 32 + lane) * 16);                  // ds_load_b128 x2
        v16h b = *(const v16h*)(Wp + ((wave * 4 + kt) * 32 + lane) * 16);      // global_load_b128 x2
        c = __builtin_amdgcn_wmma_f32_16x16x32_f16(false, a, false, b,
                                                   (short)0, c, false, false);
    }

    const int rowbase = mtile * 16 + ((lane & 16) ? 8 : 0);
    float* outp = Hout + rowbase * FOUT + col;   // int32 math; r*FOUT*4 folds into ioffset
    if (mtile * 16 + 16 <= n) {
#pragma unroll
        for (int r = 0; r < 8; ++r) outp[r * FOUT] = c[r];
    } else {
        for (int r = 0; r < 8; ++r)
            if (rowbase + r < n) outp[r * FOUT] = c[r];
    }
}

// ---------- per-node attention logits: al_s/al_d [n,HEADS]; one wave per node ----------
template <int HEADS, int HF>
__global__ void k_logits(const float* __restrict__ Hl, const float* __restrict__ a_src,
                         const float* __restrict__ a_dst, float* __restrict__ als,
                         float* __restrict__ ald, int n) {
    int node = (blockIdx.x * blockDim.x + threadIdx.x) >> 5;
    int lane = threadIdx.x & 31;
    if (node >= n) return;
    const float* row = Hl + node * HF;
    float s0 = 0.f, s1 = 0.f, d0 = 0.f, d1 = 0.f;
#pragma unroll
    for (int f0 = 0; f0 < HF; f0 += 32) {
        int f = f0 + lane;
        float v  = row[f];
        float as = a_src[f];      // (HEADS,64) flat == HF
        float ad = a_dst[f];
        if (f0 < FPH) { s0 += v * as; d0 += v * ad; }
        else          { s1 += v * as; d1 += v * ad; }
    }
#pragma unroll
    for (int off = 16; off; off >>= 1) {
        s0 += __shfl_xor(s0, off); d0 += __shfl_xor(d0, off);
        if (HEADS > 1) { s1 += __shfl_xor(s1, off); d1 += __shfl_xor(d1, off); }
    }
    if (lane == 0) {
        als[node * HEADS] = s0; ald[node * HEADS] = d0;
        if (HEADS > 1) { als[node * HEADS + 1] = s1; ald[node * HEADS + 1] = d1; }
    }
}

// ---------- zero-init out / denom / max ----------
__global__ void k_init(float* __restrict__ O, float* __restrict__ denom,
                       unsigned* __restrict__ menc, int nhf, int nh) {
    int i = blockIdx.x * blockDim.x + threadIdx.x;
    if (i < nhf) O[i] = 0.f;
    if (i < nh) { denom[i] = 0.f; menc[i] = 0u; }   // enc 0 == most negative
}

// ---------- segment max over incoming edges (monotone-uint atomicMax) ----------
template <int HEADS>
__global__ void k_edge_max(const int* __restrict__ ei, int E, int n,
                           const float* __restrict__ als, const float* __restrict__ ald,
                           unsigned* __restrict__ menc) {
    int e = blockIdx.x * blockDim.x + threadIdx.x;
    if (e >= E + n) return;
    int src, dst;
    if (e < E) { src = ei[e]; dst = ei[E + e]; } else { src = dst = e - E; }
#pragma unroll
    for (int h = 0; h < HEADS; ++h) {
        float v = lrelu(als[src * HEADS + h] + ald[dst * HEADS + h]);
        atomicMax(&menc[dst * HEADS + h], fenc(v));
    }
}

// ---------- segment sum of exp(e - max) ----------
template <int HEADS>
__global__ void k_edge_sum(const int* __restrict__ ei, int E, int n,
                           const float* __restrict__ als, const float* __restrict__ ald,
                           const unsigned* __restrict__ menc, float* __restrict__ denom) {
    int e = blockIdx.x * blockDim.x + threadIdx.x;
    if (e >= E + n) return;
    int src, dst;
    if (e < E) { src = ei[e]; dst = ei[E + e]; } else { src = dst = e - E; }
#pragma unroll
    for (int h = 0; h < HEADS; ++h) {
        float v  = lrelu(als[src * HEADS + h] + ald[dst * HEADS + h]);
        float ee = expf(v - fdec(menc[dst * HEADS + h]));
        atomicAdd(&denom[dst * HEADS + h], ee);
    }
}

// ---------- weighted scatter: out[dst] += alpha * h[src]; one wave per edge ----------
template <int HEADS, int HF>
__global__ void k_edge_aggr(const int* __restrict__ ei, int E, int n,
                            const float* __restrict__ als, const float* __restrict__ ald,
                            const unsigned* __restrict__ menc, const float* __restrict__ denom,
                            const float* __restrict__ Hl, float* __restrict__ O) {
    int wid  = (blockIdx.x * blockDim.x + threadIdx.x) >> 5;   // <= E+n < 2^31
    int lane = threadIdx.x & 31;
    if (wid >= E + n) return;
    int src, dst;
    if (wid < E) { src = ei[wid]; dst = ei[E + wid]; } else { src = dst = wid - E; }

    float alpha[HEADS];
#pragma unroll
    for (int h = 0; h < HEADS; ++h) {
        float v  = lrelu(als[src * HEADS + h] + ald[dst * HEADS + h]);
        float ee = expf(v - fdec(menc[dst * HEADS + h]));
        alpha[h] = ee / (denom[dst * HEADS + h] + 1e-16f);
    }
    const float* hrow = Hl + src * HF;    // n*HF fits int32
    float*       orow = O  + dst * HF;
    // float2 per lane: all 32 lanes active for HF=64 and HF=128; pairs never
    // straddle the 64-feature head boundary, so head = f2>>6 is exact.
#pragma unroll
    for (int f0 = 0; f0 < HF; f0 += 64) {
        int f2 = f0 + lane * 2;
        float2 hv = *(const float2*)(hrow + f2);
        float  al = alpha[f0 >> 6];
        atomicAdd(&orow[f2],     hv.x * al);
        atomicAdd(&orow[f2 + 1], hv.y * al);
    }
}

// ---------- bias + ELU (in place); HF is a power of two ----------
__global__ void k_bias_elu(float* __restrict__ O, const float* __restrict__ b,
                           int total, int hfmask) {
    int i = blockIdx.x * blockDim.x + threadIdx.x;
    if (i >= total) return;
    float v = O[i] + b[i & hfmask];
    O[i] = v > 0.f ? v : (expf(v) - 1.f);
}

// ---------- host-side layer driver ----------
template <int HEADS, int HF>
static void run_layer(const float* X, const float* W, const float* as, const float* ad,
                      const float* bias, const int* ei, int E, int n,
                      float* Hbuf, float* Obuf, float* als, float* ald,
                      unsigned* menc, float* denom, _Float16* Wp, hipStream_t stream) {
    int Etot = E + n;
    k_pack_w<HF><<<(FIN * HF + 255) / 256, 256, 0, stream>>>(W, Wp);
    k_gemm<HF><<<(n + 15) / 16, 32 * (HF / 16), 0, stream>>>(X, Wp, Hbuf, n);
    k_logits<HEADS, HF><<<(n + 7) / 8, 256, 0, stream>>>(Hbuf, as, ad, als, ald, n);
    k_init<<<(n * HF + 255) / 256, 256, 0, stream>>>(Obuf, denom, menc, n * HF, n * HEADS);
    k_edge_max<HEADS><<<(Etot + 255) / 256, 256, 0, stream>>>(ei, E, n, als, ald, menc);
    k_edge_sum<HEADS><<<(Etot + 255) / 256, 256, 0, stream>>>(ei, E, n, als, ald, menc, denom);
    k_edge_aggr<HEADS, HF><<<(Etot + 7) / 8, 256, 0, stream>>>(ei, E, n, als, ald, menc,
                                                               denom, Hbuf, Obuf);
    k_bias_elu<<<(n * HF + 255) / 256, 256, 0, stream>>>(Obuf, bias, n * HF, HF - 1);
}

extern "C" void kernel_launch(void* const* d_in, const int* in_sizes, int n_in,
                              void* d_out, int out_size, void* d_ws, size_t ws_size,
                              hipStream_t stream) {
    const float* x   = (const float*)d_in[0];
    const float* W1  = (const float*)d_in[1];
    const float* as1 = (const float*)d_in[2];
    const float* ad1 = (const float*)d_in[3];
    const float* b1  = (const float*)d_in[4];
    const float* W2  = (const float*)d_in[5];
    const float* as2 = (const float*)d_in[6];
    const float* ad2 = (const float*)d_in[7];
    const float* b2  = (const float*)d_in[8];
    const float* W3  = (const float*)d_in[9];
    const float* as3 = (const float*)d_in[10];
    const float* ad3 = (const float*)d_in[11];
    const float* b3  = (const float*)d_in[12];
    const int*   ei  = (const int*)d_in[13];

    const int n = in_sizes[0] / FIN;       // 50000
    const int E = in_sizes[13] / 2;        // 1600000

    // workspace carve (256B aligned)
    char* p = (char*)d_ws;
    auto carve = [&](size_t bytes) {
        char* q = p;
        p += (bytes + 255) & ~(size_t)255;
        return (void*)q;
    };
    float*     Hbuf  = (float*)carve((size_t)n * FIN * sizeof(float));   // linear output
    float*     Xbuf  = (float*)carve((size_t)n * FIN * sizeof(float));   // layer I/O
    float*     als   = (float*)carve((size_t)n * 2 * sizeof(float));
    float*     ald   = (float*)carve((size_t)n * 2 * sizeof(float));
    unsigned*  menc  = (unsigned*)carve((size_t)n * 2 * sizeof(unsigned));
    float*     denom = (float*)carve((size_t)n * 2 * sizeof(float));
    _Float16*  Wp    = (_Float16*)carve((size_t)FIN * FIN * sizeof(_Float16));

    // layer 1: x -> Xbuf          (heads=2, hf=128)
    run_layer<2, 128>(x, W1, as1, ad1, b1, ei, E, n,
                      Hbuf, Xbuf, als, ald, menc, denom, Wp, stream);
    // layer 2: Xbuf -> Xbuf (GEMM consumes Xbuf before k_init zeroes it; stream-ordered)
    run_layer<2, 128>(Xbuf, W2, as2, ad2, b2, ei, E, n,
                      Hbuf, Xbuf, als, ald, menc, denom, Wp, stream);
    // layer 3: Xbuf -> d_out      (heads=1, hf=64)
    run_layer<1, 64>(Xbuf, W3, as3, ad3, b3, ei, E, n,
                     Hbuf, (float*)d_out, als, ald, menc, denom, Wp, stream);
}
